// AdaptiveCausalAttention_54443005444877
// MI455X (gfx1250) — compile-verified, matrix operand order checked by reading, not run on hardware
//
#include <hip/hip_runtime.h>
#include <hip/hip_bf16.h>
#include <math.h>

// ---------------- problem constants ----------------
#define NB 4
#define TQ 1024
#define CC 1024
#define HH 16
#define HD 64
#define NQKV (3 * CC)

typedef __attribute__((ext_vector_type(16))) _Float16 v16h;
typedef __attribute__((ext_vector_type(8)))  float    v8f;

__device__ __forceinline__ v8f wmma_f16(v16h a, v16h b, v8f c) {
    return __builtin_amdgcn_wmma_f32_16x16x32_f16(
        false, a, false, b, (short)0, c, false, false);
}

// Per-lane 16-half fragment load. Lane l: row (l&15); K chunks
// [c0..c0+7] and [c0+16..c0+23], c0=(l>>4)*8. Two 16B ds/global b128 loads.
__device__ __forceinline__ v16h load_frag16(const _Float16* p, int c0) {
    union { uint4 u[2]; v16h h; } r;
    r.u[0] = *(const uint4*)(p + c0);
    r.u[1] = *(const uint4*)(p + c0 + 16);
    return r.h;
}

// CDNA5 async global->LDS copy (ASYNCcnt-tracked, bypasses VGPRs).
__device__ __forceinline__ void async_b128(const _Float16* lds_dst, const _Float16* gsrc) {
    unsigned lds_off = (unsigned)(size_t)lds_dst;   // low 32 bits of LDS aperture = LDS offset
    asm volatile("global_load_async_to_lds_b128 %0, %1, off"
                 :: "v"(lds_off), "v"((const void*)gsrc) : "memory");
}
__device__ __forceinline__ void wait_async0() {
    asm volatile("s_wait_asynccnt 0" ::: "memory");
}

__device__ __forceinline__ uint4 pack8h_f(const float* f) {
    union { _Float16 h[8]; uint4 u; } r;
    #pragma unroll
    for (int i = 0; i < 8; ++i) r.h[i] = (_Float16)f[i];
    return r.u;
}

// =====================================================================
// Preprocess kernels (one-shot, bandwidth-trivial vs the GEMMs)
// =====================================================================
// f32 -> f16 elementwise (x)
__global__ __launch_bounds__(256)
void cvt_f16_kernel(const float* __restrict__ src, _Float16* __restrict__ dst)
{
    size_t i = ((size_t)blockIdx.x * 256 + threadIdx.x) * 4;
    float4 f = *(const float4*)(src + i);
    union { _Float16 h[4]; uint2 u; } r;
    r.h[0] = (_Float16)f.x; r.h[1] = (_Float16)f.y;
    r.h[2] = (_Float16)f.z; r.h[3] = (_Float16)f.w;
    *(uint2*)(dst + i) = r.u;
}

// W[k][n] f32 -> Wt[n][k] f16 (32x32 LDS tile transpose)
__global__ __launch_bounds__(256)
void transpose_cvt_kernel(const float* __restrict__ src, _Float16* __restrict__ dst,
                          int K, int N)
{
    __shared__ float tile[32][33];
    const int k0 = blockIdx.x * 32;
    const int n0 = blockIdx.y * 32;
    const int tx = threadIdx.x & 31;
    const int ty = threadIdx.x >> 5;     // 0..7
    #pragma unroll
    for (int r = 0; r < 32; r += 8)
        tile[ty + r][tx] = src[(size_t)(k0 + ty + r) * N + n0 + tx];
    __syncthreads();
    #pragma unroll
    for (int r = 0; r < 32; r += 8)
        dst[(size_t)(n0 + ty + r) * K + k0 + tx] = (_Float16)tile[tx][ty + r];
}

// =====================================================================
// Generic 128x128 f16 GEMM body: async-staged, double-buffered LDS,
// one barrier per 32-k-step, 8 WMMA per wave per step.
// A: [M][K] f16 row-major; Bt: [N][K] f16 (i.e. B^T row-major).
// =====================================================================
struct GemmAccum { v8f a[2][4]; };

template <int KDIM>
__device__ __forceinline__ void gemm_core(const _Float16* __restrict__ A,
                                          const _Float16* __restrict__ Bt,
                                          _Float16* AsBuf, _Float16* BtBuf,
                                          int m0, int n0, GemmAccum& out)
{
    const int tid  = threadIdx.x;
    const int lane = tid & 31;
    const int wid  = tid >> 5;
    const int l16  = lane & 15;
    const int c0   = (lane >> 4) * 8;
    const int wm   = (wid & 3) * 32;
    const int wn   = (wid >> 2) * 64;

    // staging coords: each thread owns one 16-half row segment in A and Bt
    const int ar = tid >> 1;             // 0..127
    const int ac = (tid & 1) * 16;       // 0 / 16 halves

    #pragma unroll
    for (int i = 0; i < 2; ++i)
        #pragma unroll
        for (int j = 0; j < 4; ++j) out.a[i][j] = (v8f){};

    // prologue: stage k0 = 0 into buffer 0
    {
        const _Float16* as = A  + (size_t)(m0 + ar) * KDIM + ac;
        const _Float16* bs = Bt + (size_t)(n0 + ar) * KDIM + ac;
        async_b128(&AsBuf[ar * 48 + ac],     as);
        async_b128(&AsBuf[ar * 48 + ac + 8], as + 8);
        async_b128(&BtBuf[ar * 48 + ac],     bs);
        async_b128(&BtBuf[ar * 48 + ac + 8], bs + 8);
    }
    wait_async0();
    __syncthreads();

    for (int k0 = 0, it = 0; k0 < KDIM; k0 += 32, ++it) {
        const int cur = (it & 1) * (128 * 48);
        const int nxt = ((it & 1) ^ 1) * (128 * 48);
        if (k0 + 32 < KDIM) {   // stage next tile asynchronously
            const _Float16* as = A  + (size_t)(m0 + ar) * KDIM + k0 + 32 + ac;
            const _Float16* bs = Bt + (size_t)(n0 + ar) * KDIM + k0 + 32 + ac;
            async_b128(&AsBuf[nxt + ar * 48 + ac],     as);
            async_b128(&AsBuf[nxt + ar * 48 + ac + 8], as + 8);
            async_b128(&BtBuf[nxt + ar * 48 + ac],     bs);
            async_b128(&BtBuf[nxt + ar * 48 + ac + 8], bs + 8);
        }
        v16h a0 = load_frag16(&AsBuf[cur + (wm + l16) * 48], c0);
        v16h a1 = load_frag16(&AsBuf[cur + (wm + 16 + l16) * 48], c0);
        v16h b0 = load_frag16(&BtBuf[cur + (wn + l16) * 48], c0);
        v16h b1 = load_frag16(&BtBuf[cur + (wn + 16 + l16) * 48], c0);
        v16h b2 = load_frag16(&BtBuf[cur + (wn + 32 + l16) * 48], c0);
        v16h b3 = load_frag16(&BtBuf[cur + (wn + 48 + l16) * 48], c0);
        out.a[0][0] = wmma_f16(a0, b0, out.a[0][0]);
        out.a[0][1] = wmma_f16(a0, b1, out.a[0][1]);
        out.a[0][2] = wmma_f16(a0, b2, out.a[0][2]);
        out.a[0][3] = wmma_f16(a0, b3, out.a[0][3]);
        out.a[1][0] = wmma_f16(a1, b0, out.a[1][0]);
        out.a[1][1] = wmma_f16(a1, b1, out.a[1][1]);
        out.a[1][2] = wmma_f16(a1, b2, out.a[1][2]);
        out.a[1][3] = wmma_f16(a1, b3, out.a[1][3]);
        wait_async0();
        __syncthreads();
    }
}

// =====================================================================
// Kernel 1: QKV GEMM (xh[4096,1024] f16) @ (WqkvT[3072,1024] f16)
// -> q/k in [b,h,t,d] f16; v written TRANSPOSED as [b,h,d,t] f16.
// =====================================================================
__global__ __launch_bounds__(256)
void qkv_gemm_kernel(const _Float16* __restrict__ xh, const _Float16* __restrict__ wqkvt,
                     _Float16* __restrict__ qb, _Float16* __restrict__ kb,
                     _Float16* __restrict__ vtg)
{
    __shared__ __align__(16) _Float16 AsBuf[2 * 128 * 48];
    __shared__ __align__(16) _Float16 BtBuf[2 * 128 * 48];

    const int tid  = threadIdx.x;
    const int lane = tid & 31;
    const int wid  = tid >> 5;
    const int l16  = lane & 15;
    const int hl   = lane >> 4;
    const int m0   = blockIdx.x * 128;
    const int n0   = blockIdx.y * 128;
    const int wm   = (wid & 3) * 32;
    const int wn   = (wid >> 2) * 64;

    GemmAccum acc;
    gemm_core<CC>(xh, wqkvt, AsBuf, BtBuf, m0, n0, acc);

    #pragma unroll
    for (int mt = 0; mt < 2; ++mt)
        #pragma unroll
        for (int f = 0; f < 4; ++f) {
            int col   = n0 + wn + f * 16 + l16;
            int sel   = col >> 10;                 // 0=q 1=k 2=v (block-uniform)
            int w1024 = col & 1023;
            int h = w1024 >> 6;
            int d = w1024 & 63;
            int row0 = m0 + wm + mt * 16 + hl * 8; // rows row0..row0+7
            int b    = row0 >> 10;
            int t0   = row0 & 1023;
            if (sel == 2) {
                // V transposed [b,h,d,t]: 8 consecutive t -> one b128 store
                float fv[8];
                #pragma unroll
                for (int v = 0; v < 8; ++v) fv[v] = acc.a[mt][f][v];
                uint4 pk = pack8h_f(fv);
                *(uint4*)&vtg[((size_t)(b * HH + h) * HD + d) * TQ + t0] = pk;
            } else {
                _Float16* dst = (sel == 0) ? qb : kb;
                #pragma unroll
                for (int v = 0; v < 8; ++v)
                    dst[(((size_t)(b * HH + h)) * TQ + t0 + v) * HD + d] =
                        (_Float16)acc.a[mt][f][v];
            }
        }
}

// =====================================================================
// Kernel 2: adaptive-span causal attention.
// 256 query rows / workgroup, 32 rows / wave, 32-key steps,
// V^T tiles async-staged + double-buffered, 16 WMMA per wave per step.
// =====================================================================
__global__ __launch_bounds__(256)
void attn_kernel(const _Float16* __restrict__ qb, const _Float16* __restrict__ kb,
                 const _Float16* __restrict__ vtg,
                 const float* __restrict__ span_params,
                 const float* __restrict__ stride_params,
                 _Float16* __restrict__ yb)
{
    const int bh = blockIdx.y;
    const int h  = bh & (HH - 1);
    const int b  = bh >> 4;
    const int qgroup = blockIdx.x * 256;

    const int tid  = threadIdx.x;
    const int lane = tid & 31;
    const int wid  = tid >> 5;
    const int l16  = lane & 15;
    const int hl   = lane >> 4;
    const int c0   = hl * 8;
    const int qbase = qgroup + wid * 32;

    const float sp  = 1024.f / (1.f + __expf(-span_params[h]));
    const float s0  = stride_params[h * 2 + 0];
    const float s1  = stride_params[h * 2 + 1];
    const float mx  = fmaxf(s0, s1);
    const float e0  = __expf(s0 - mx), e1 = __expf(s1 - mx);
    const float sw0 = e0 / (e0 + e1),  sw1 = e1 / (e0 + e1);

    const _Float16* Q   = qb  + (size_t)bh * TQ * HD;
    const _Float16* Kp  = kb  + (size_t)bh * TQ * HD;
    const _Float16* Vtg = vtg + (size_t)bh * HD * TQ;   // [d][t]

    __shared__ __align__(16) _Float16 Vt[2][64 * 48];   // staged V^T tile (32 keys)
    __shared__ __align__(16) _Float16 Pb[8][32 * 48];   // per-wave P

    // V staging coords: thread copies one 16B chunk: d = tid>>2, chunk (tid&3)*8
    const int vd = tid >> 2;
    const int vc = (tid & 3) * 8;

    // Q fragments [qtile][dpart]
    v16h qf[2][2];
    #pragma unroll
    for (int t = 0; t < 2; ++t) {
        const _Float16* qrow = Q + (size_t)(qbase + t * 16 + l16) * HD;
        qf[t][0] = load_frag16(qrow,      c0);
        qf[t][1] = load_frag16(qrow + 32, c0);
    }

    v8f y[2][4];
    #pragma unroll
    for (int t = 0; t < 2; ++t)
        #pragma unroll
        for (int f = 0; f < 4; ++f) y[t][f] = (v8f){};
    v8f rs[2] = {(v8f){}, (v8f){}};

    const int jend = (qgroup + 256 < TQ) ? (qgroup + 256) : TQ;

    // prologue: stage first V tile
    async_b128(&Vt[0][vd * 48 + vc], Vtg + (size_t)vd * TQ + 0 + vc);
    wait_async0();
    __syncthreads();

    for (int j = 0, it = 0; j < jend; j += 32, ++it) {
        const int cur = it & 1;
        const int nxt = cur ^ 1;
        if (j + 32 < jend)
            async_b128(&Vt[nxt][vd * 48 + vc], Vtg + (size_t)vd * TQ + (j + 32) + vc);

        if (j <= qbase + 31) {                 // wave-uniform causal skip
            v16h kf[2][2];
            #pragma unroll
            for (int kt = 0; kt < 2; ++kt) {
                const _Float16* krow = Kp + (size_t)(j + kt * 16 + l16) * HD;
                kf[kt][0] = load_frag16(krow,      c0);
                kf[kt][1] = load_frag16(krow + 32, c0);
            }
            v8f s[2][2];
            #pragma unroll
            for (int t = 0; t < 2; ++t)
                #pragma unroll
                for (int kt = 0; kt < 2; ++kt) {
                    v8f z = (v8f){};
                    z = wmma_f16(qf[t][0], kf[kt][0], z);
                    z = wmma_f16(qf[t][1], kf[kt][1], z);
                    s[t][kt] = z;
                }
            _Float16* P = Pb[wid];
            #pragma unroll
            for (int t = 0; t < 2; ++t)
                #pragma unroll
                for (int kt = 0; kt < 2; ++kt) {
                    int jcol = j + kt * 16 + l16;
                    #pragma unroll
                    for (int v = 0; v < 8; ++v) {
                        int irow = qbase + t * 16 + v + hl * 8;
                        int rel  = irow - jcol;
                        float p  = 0.f;
                        if (rel >= 0) {
                            float clipm = fminf(fmaxf((32.f + sp - (float)rel) * (1.f / 32.f), 0.f), 1.f);
                            float w     = sw0 + (((rel & 1) == 0) ? sw1 : 0.f);
                            p = __expf(0.125f * s[t][kt][v]) * clipm * w;
                        }
                        rs[t][v] += p;
                        P[(t * 16 + v + hl * 8) * 48 + kt * 16 + l16] = (_Float16)p;
                    }
                }
            asm volatile("s_wait_dscnt 0" ::: "memory");

            v16h vf[4];
            #pragma unroll
            for (int f = 0; f < 4; ++f)
                vf[f] = load_frag16(&Vt[cur][(f * 16 + l16) * 48], c0);
            #pragma unroll
            for (int t = 0; t < 2; ++t) {
                v16h pf = load_frag16(&P[(t * 16 + l16) * 48], c0);
                #pragma unroll
                for (int f = 0; f < 4; ++f)
                    y[t][f] = wmma_f16(pf, vf[f], y[t][f]);
            }
        }
        wait_async0();
        __syncthreads();
    }

    #pragma unroll
    for (int t = 0; t < 2; ++t)
        #pragma unroll
        for (int m = 1; m <= 8; m <<= 1) {
            #pragma unroll
            for (int v = 0; v < 8; ++v) rs[t][v] += __shfl_xor(rs[t][v], m, 32);
        }

    #pragma unroll
    for (int t = 0; t < 2; ++t)
        #pragma unroll
        for (int f = 0; f < 4; ++f) {
            int dcol = f * 16 + l16;
            #pragma unroll
            for (int v = 0; v < 8; ++v) {
                int trow = qbase + t * 16 + v + hl * 8;
                float val = y[t][f][v] / rs[t][v];
                yb[((size_t)b * TQ + trow) * CC + h * HD + dcol] = (_Float16)val;
            }
        }
}

// =====================================================================
// Kernel 3: projection GEMM (y[4096,1024] f16) @ (WprojT[1024,1024] f16) -> f32
// =====================================================================
__global__ __launch_bounds__(256)
void proj_gemm_kernel(const _Float16* __restrict__ ya, const _Float16* __restrict__ wprojt,
                      float* __restrict__ out)
{
    __shared__ __align__(16) _Float16 AsBuf[2 * 128 * 48];
    __shared__ __align__(16) _Float16 BtBuf[2 * 128 * 48];

    const int tid  = threadIdx.x;
    const int lane = tid & 31;
    const int wid  = tid >> 5;
    const int l16  = lane & 15;
    const int hl   = lane >> 4;
    const int m0   = blockIdx.x * 128;
    const int n0   = blockIdx.y * 128;
    const int wm   = (wid & 3) * 32;
    const int wn   = (wid >> 2) * 64;

    GemmAccum acc;
    gemm_core<CC>(ya, wprojt, AsBuf, BtBuf, m0, n0, acc);

    #pragma unroll
    for (int mt = 0; mt < 2; ++mt)
        #pragma unroll
        for (int f = 0; f < 4; ++f) {
            int col = n0 + wn + f * 16 + l16;
            #pragma unroll
            for (int v = 0; v < 8; ++v) {
                int row = m0 + wm + mt * 16 + v + hl * 8;
                out[(size_t)row * CC + col] = acc.a[mt][f][v];
            }
        }
}

// =====================================================================
// Kernel 4: span loss scalar -> out[B*T*C]
// =====================================================================
__global__ void span_loss_kernel(const float* __restrict__ span_params, float* __restrict__ out)
{
    if (threadIdx.x == 0 && blockIdx.x == 0) {
        float s = 0.f;
        #pragma unroll
        for (int i = 0; i < HH; ++i)
            s += 1024.f / (1.f + __expf(-span_params[i]));
        out[(size_t)NB * TQ * CC] = 2e-6f * s / (float)HH;
    }
}

// =====================================================================
extern "C" void kernel_launch(void* const* d_in, const int* in_sizes, int n_in,
                              void* d_out, int out_size, void* d_ws, size_t ws_size,
                              hipStream_t stream) {
    (void)in_sizes; (void)n_in; (void)out_size; (void)ws_size;
    const float* x        = (const float*)d_in[0];
    const float* Wqkv     = (const float*)d_in[1];
    const float* Wproj    = (const float*)d_in[2];
    const float* span_p   = (const float*)d_in[3];
    const float* stride_p = (const float*)d_in[4];
    float* out = (float*)d_out;

    // workspace layout (f16 elements)
    _Float16* xh     = (_Float16*)d_ws;                       // 4096*1024
    _Float16* wqkvt  = xh     + (size_t)NB * TQ * CC;         // 3072*1024
    _Float16* wprojt = wqkvt  + (size_t)NQKV * CC;            // 1024*1024
    _Float16* qb     = wprojt + (size_t)CC * CC;              // 4M
    _Float16* kb     = qb     + (size_t)NB * HH * TQ * HD;
    _Float16* vtg    = kb     + (size_t)NB * HH * TQ * HD;    // [b,h,d,t]
    _Float16* yb     = vtg    + (size_t)NB * HH * TQ * HD;

    cvt_f16_kernel<<<(NB * TQ * CC) / 1024, 256, 0, stream>>>(x, xh);
    transpose_cvt_kernel<<<dim3(CC / 32, NQKV / 32), 256, 0, stream>>>(Wqkv, wqkvt, CC, NQKV);
    transpose_cvt_kernel<<<dim3(CC / 32, CC / 32), 256, 0, stream>>>(Wproj, wprojt, CC, CC);

    qkv_gemm_kernel<<<dim3((NB * TQ) / 128, NQKV / 128), 256, 0, stream>>>(xh, wqkvt, qb, kb, vtg);
    attn_kernel<<<dim3(TQ / 256, NB * HH), 256, 0, stream>>>(qb, kb, vtg, span_p, stride_p, yb);
    proj_gemm_kernel<<<dim3((NB * TQ) / 128, CC / 128), 256, 0, stream>>>(yb, wprojt, out);
    span_loss_kernel<<<1, 32, 0, stream>>>(span_p, out);
}
